// BahadanauAttention_30666066494188
// MI455X (gfx1250) — compile-verified, matrix operand order checked by reading, not run on hardware
//
#include <hip/hip_runtime.h>
#include <hip/hip_bf16.h>
#include <math.h>

typedef __attribute__((ext_vector_type(16))) __bf16 v16bf;
typedef __attribute__((ext_vector_type(8)))  float  v8f;

#define BB   32
#define SS   2048
#define HH   1024
#define K2H  2048   // 2*H : inner dim of enc rows and Wk rows

// ---------------- workspace layout (bytes) ----------------
// [0,           4 MB)   Wk as bf16, row-major [H][2H]
// [4 MB,        +128KB) query f32 [B][H]
// [.. ,         +256KB) scores f32 [B][S]
// [.. ,         +256KB) alphas f32 [B][S]
#define WS_WKBF   0
#define WS_QUERY  (HH * K2H * 2)                 // 4194304
#define WS_SCORES (WS_QUERY + BB * HH * 4)       // +131072
#define WS_ALPHA  (WS_SCORES + BB * SS * 4)      // +262144

// ---------------- kernel 1: Wk fp32 -> bf16 ----------------
__global__ __launch_bounds__(256)
void wk_to_bf16_kernel(const float* __restrict__ wk, __bf16* __restrict__ out) {
    int i = (blockIdx.x * 256 + threadIdx.x) * 8;
    float4 a = *(const float4*)(wk + i);
    float4 b = *(const float4*)(wk + i + 4);
    union { __bf16 h[8]; uint4 u; } r;
    r.h[0] = (__bf16)a.x; r.h[1] = (__bf16)a.y; r.h[2] = (__bf16)a.z; r.h[3] = (__bf16)a.w;
    r.h[4] = (__bf16)b.x; r.h[5] = (__bf16)b.y; r.h[6] = (__bf16)b.z; r.h[7] = (__bf16)b.w;
    *(uint4*)(out + i) = r.u;
}

// ---------------- kernel 2: query = dec @ Wq^T ----------------
__global__ __launch_bounds__(256)
void query_kernel(const float* __restrict__ dec, const float* __restrict__ wq,
                  float* __restrict__ q) {
    int b = blockIdx.y;
    int h = blockIdx.x * 256 + threadIdx.x;
    const float* d = dec + b * HH;
    const float* w = wq + (size_t)h * HH;
    float acc = 0.f;
#pragma unroll 4
    for (int e = 0; e < HH; e += 4) {
        float4 dv = *(const float4*)(d + e);
        float4 wv = *(const float4*)(w + e);
        acc += dv.x * wv.x + dv.y * wv.y + dv.z * wv.z + dv.w * wv.w;
    }
    q[b * HH + h] = acc;
}

// ---------------- kernel 3: fused key-GEMM + tanh + dot(We) ----------------
// One WG = 32 rows of (b*S+s), split as 2 row-groups of 16.
// 8 waves x 128 columns of H each (8 WMMA col-tiles); each B fragment feeds
// 2 WMMAs (one per row-group) -> 1 global_load_b128 per WMMA.
__global__ __launch_bounds__(256)
void scores_kernel(const float* __restrict__ enc, const __bf16* __restrict__ wkb,
                   const float* __restrict__ query, const float* __restrict__ we,
                   float* __restrict__ scores) {
    __shared__ __bf16 ldsA[32 * 40];          // 32 rows x 32 K, padded stride 40
    __shared__ float  ldsR[8][32][16];        // wave x lane x (group*8 + vgpr)

    const int tid  = threadIdx.x;
    const int wave = tid >> 5;
    const int lane = tid & 31;
    const int row0 = blockIdx.x * 32;         // flat (b*S + s) row base
    const int bIdx = row0 / SS;               // 32-row tile never crosses a batch
    const int h0   = wave * 128;              // this wave's column base in H

    const int nsub = (lane >= 16) ? (lane - 16) : lane;  // row (A) / col (B) within tile
    const int kb   = (lane >= 16) ? 8 : 0;               // K sub-block per ISA layout

    v8f acc[2][8] = {};

    for (int k0 = 0; k0 < K2H; k0 += 32) {
        __syncthreads();
        // stage A tile: 32 rows x 32 k = 1024 floats -> bf16 in LDS, 4 per thread
        {
            int idx = tid * 4;
            int r = idx >> 5;
            int k = idx & 31;
            float4 v = *(const float4*)(enc + (size_t)(row0 + r) * K2H + k0 + k);
            union { __bf16 h[4]; uint2 u; } pk;
            pk.h[0] = (__bf16)v.x; pk.h[1] = (__bf16)v.y;
            pk.h[2] = (__bf16)v.z; pk.h[3] = (__bf16)v.w;
            *(uint2*)&ldsA[r * 40 + k] = pk.u;
        }
        __syncthreads();

        // A fragments for both 16-row groups:
        // lanes 0-15 row=lane K{0-7,16-23}; lanes 16-31 row=lane-16 K{8-15,24-31}
        union { v16bf v; uint4 q[2]; } afr[2];
#pragma unroll
        for (int g = 0; g < 2; g++) {
            const __bf16* ap = &ldsA[(g * 16 + nsub) * 40 + kb];
            afr[g].q[0] = *(const uint4*)(ap);
            afr[g].q[1] = *(const uint4*)(ap + 16);
        }

#pragma unroll
        for (int t = 0; t < 8; t++) {
            int h = h0 + t * 16 + nsub;
            const __bf16* bp = wkb + (size_t)h * K2H + k0 + kb;  // B[k][n] = Wk[h][e]
            union { v16bf v; uint4 q[2]; } bfr;
            bfr.q[0] = *(const uint4*)(bp);
            bfr.q[1] = *(const uint4*)(bp + 16);
            acc[0][t] = __builtin_amdgcn_wmma_f32_16x16x32_bf16(
                false, afr[0].v, false, bfr.v, (short)0, acc[0][t], false, false);
            acc[1][t] = __builtin_amdgcn_wmma_f32_16x16x32_bf16(
                false, afr[1].v, false, bfr.v, (short)0, acc[1][t], false, false);
        }
    }

    // epilogue: p[g][v] = sum_tiles tanh(key + query[b,h]) * We[h]
    float p[2][8];
#pragma unroll
    for (int g = 0; g < 2; g++)
#pragma unroll
        for (int v = 0; v < 8; v++) p[g][v] = 0.f;

#pragma unroll
    for (int t = 0; t < 8; t++) {
        int h = h0 + t * 16 + nsub;
        float qv = query[bIdx * HH + h];
        float wv = we[h];
#pragma unroll
        for (int g = 0; g < 2; g++)
#pragma unroll
            for (int v = 0; v < 8; v++)
                p[g][v] += tanhf(acc[g][t][v] + qv) * wv;
    }
#pragma unroll
    for (int g = 0; g < 2; g++)
#pragma unroll
        for (int v = 0; v < 8; v++) ldsR[wave][lane][g * 8 + v] = p[g][v];
    __syncthreads();

    // deterministic tree: thread m (0..31) sums row m across 8 waves x 16 lanes
    if (tid < 32) {
        int m  = tid;
        int g  = m >> 4;             // row-group
        int mm = m & 15;             // row within group
        int v  = mm & 7;
        int lb = (mm < 8) ? 0 : 16;  // lanes 0-15 hold M=v, lanes 16-31 hold M=v+8
        float s = 0.f;
        for (int w = 0; w < 8; w++)
            for (int l = 0; l < 16; l++)
                s += ldsR[w][lb + l][g * 8 + v];
        scores[row0 + m] = s;
    }
}

// ---------------- kernel 4: softmax over S per batch ----------------
__global__ __launch_bounds__(256)
void softmax_kernel(const float* __restrict__ scores, float* __restrict__ alpha) {
    __shared__ float red[256];
    int b = blockIdx.x;
    int tid = threadIdx.x;
    const float* sc = scores + b * SS;
    float* al = alpha + b * SS;

    float local[8];
    float mx = -INFINITY;
#pragma unroll
    for (int i = 0; i < 8; i++) { local[i] = sc[tid + i * 256]; mx = fmaxf(mx, local[i]); }
    red[tid] = mx; __syncthreads();
    for (int off = 128; off > 0; off >>= 1) {
        if (tid < off) red[tid] = fmaxf(red[tid], red[tid + off]);
        __syncthreads();
    }
    mx = red[0]; __syncthreads();

    float sum = 0.f;
#pragma unroll
    for (int i = 0; i < 8; i++) { local[i] = __expf(local[i] - mx); sum += local[i]; }
    red[tid] = sum; __syncthreads();
    for (int off = 128; off > 0; off >>= 1) {
        if (tid < off) red[tid] += red[tid + off];
        __syncthreads();
    }
    float inv = 1.f / red[0];
#pragma unroll
    for (int i = 0; i < 8; i++) al[tid + i * 256] = local[i] * inv;
}

// ---------------- kernel 5: context = alpha @ enc ----------------
__global__ __launch_bounds__(256)
void context_kernel(const float* __restrict__ enc, const float* __restrict__ alpha,
                    float* __restrict__ out) {
    int b = blockIdx.y;
    int e0 = blockIdx.x * 512 + threadIdx.x * 2;
    const float* al = alpha + b * SS;
    const float* ep = enc + (size_t)b * SS * K2H + e0;
    float ax = 0.f, ay = 0.f;
#pragma unroll 8
    for (int s = 0; s < SS; s++) {
        float a = al[s];
        float2 v = *(const float2*)(ep + (size_t)s * K2H);
        ax += a * v.x; ay += a * v.y;
    }
    float* o = out + b * K2H + e0;
    o[0] = ax; o[1] = ay;
}

// ---------------- launch ----------------
extern "C" void kernel_launch(void* const* d_in, const int* in_sizes, int n_in,
                              void* d_out, int out_size, void* d_ws, size_t ws_size,
                              hipStream_t stream) {
    const float* enc = (const float*)d_in[0];   // [B,S,2H]
    const float* dec = (const float*)d_in[1];   // [B,1,H]
    const float* wq  = (const float*)d_in[2];   // [H,H]
    const float* wk  = (const float*)d_in[3];   // [H,2H]
    const float* we  = (const float*)d_in[4];   // [1,H]
    float* out = (float*)d_out;                 // [B,1,2H]

    char* ws = (char*)d_ws;
    __bf16* wkb   = (__bf16*)(ws + WS_WKBF);
    float*  query = (float*)(ws + WS_QUERY);
    float*  score = (float*)(ws + WS_SCORES);
    float*  alpha = (float*)(ws + WS_ALPHA);

    wk_to_bf16_kernel<<<(HH * K2H) / (256 * 8), 256, 0, stream>>>(wk, wkb);
    query_kernel<<<dim3(HH / 256, BB), 256, 0, stream>>>(dec, wq, query);
    scores_kernel<<<(BB * SS) / 32, 256, 0, stream>>>(enc, wkb, query, we, score);
    softmax_kernel<<<BB, 256, 0, stream>>>(score, alpha);
    context_kernel<<<dim3(K2H / 512, BB), 256, 0, stream>>>(enc, alpha, out);
}
